// MyCRF_6158983102709
// MI455X (gfx1250) — compile-verified, matrix operand order checked by reading, not run on hardware
//
#include <hip/hip_runtime.h>
#include <hip/hip_bf16.h>
#include <cstddef>

#define NUM_TAGS 256
#define SEQ      512
#define BATCH    64
#define BPB      16          // batch chains per workgroup (WMMA M dimension)
#define FWD_THREADS 512      // 16 wave32s: one N-tile of 16 tags per wave

typedef __attribute__((ext_vector_type(16))) _Float16 v16h;
typedef __attribute__((ext_vector_type(8)))  float    v8f;

// ---------------------------------------------------------------------------
// Forward (partition function) kernel: 4 blocks x 16 chains.
// logsumexp recurrence rewritten as f16 WMMA matmul with f32 accumulate:
//   alpha'[b,n] = logit[b,t,n] + M[b] + log( sum_k exp(alpha[b,k]-M[b]) * exp(T[k,n]) )
// exp(T) is pinned in VGPRs (8 x v16h per wave) for all 511 steps.
// ---------------------------------------------------------------------------
__global__ __launch_bounds__(FWD_THREADS, 1)
void crf_forward_kernel(const float* __restrict__ logits,
                        const int*   __restrict__ mask,
                        const float* __restrict__ transitions,
                        float*       __restrict__ log_den)
{
    __shared__ float s_alpha[BPB][NUM_TAGS];   // 16 KB, current alpha (f32)
    __shared__ v16h  s_A[8][32];               // 8 KB, A operand packed per k-chunk
    __shared__ float s_scr[BPB][16];           // reduction scratch
    __shared__ float s_M[BPB];                 // per-chain running max

    const int tid  = threadIdx.x;
    const int lane = tid & 31;
    const int wave = tid >> 5;                 // 0..15 == N-tile index
    const int nIdx = lane & 15;
    const int lh   = lane >> 4;                // lane half (operand layout)
    const int bb   = blockIdx.x * BPB;         // batch base

    // ---- pack B = exp(transitions) into registers, WMMA 16-bit B layout ----
    v16h Bt[8];
    {
        const int n = wave * 16 + nIdx;
        #pragma unroll
        for (int kc = 0; kc < 8; ++kc) {
            v16h v;
            #pragma unroll
            for (int e = 0; e < 16; ++e) {
                const int k = kc * 32 + (e & 7) + ((e >> 3) << 4) + lh * 8;
                v[e] = (_Float16)__expf(transitions[k * NUM_TAGS + n]);
            }
            Bt[kc] = v;
        }
    }

    // ---- alpha0 = logits[:, 0, :] ----
    for (int i = tid; i < BPB * NUM_TAGS; i += FWD_THREADS) {
        const int b = i >> 8, j = i & 255;
        s_alpha[b][j] = logits[(size_t)(bb + b) * SEQ * NUM_TAGS + j];
    }
    __syncthreads();

    for (int t = 1; t < SEQ; ++t) {
        // -- phase 1: per-chain max (numerical rescale) --
        if (tid < 256) {
            const int b = tid >> 4, j0 = (tid & 15) << 4;
            float m = s_alpha[b][j0];
            #pragma unroll
            for (int j = 1; j < 16; ++j) m = fmaxf(m, s_alpha[b][j0 + j]);
            s_scr[b][tid & 15] = m;
        }
        __syncthreads();
        if (tid < BPB) {
            float m = s_scr[tid][0];
            #pragma unroll
            for (int j = 1; j < 16; ++j) m = fmaxf(m, s_scr[tid][j]);
            s_M[tid] = m;
        }
        __syncthreads();
        // -- phase 2: pack A = exp(alpha - M) f16 in WMMA A layout --
        if (tid < 256) {
            const int kc = tid >> 5;
            const int bA = lane & 15;
            const float mb = s_M[bA];
            _Float16* ah = (_Float16*)&s_A[kc][lane];
            #pragma unroll
            for (int e = 0; e < 16; ++e) {
                const int k = kc * 32 + (e & 7) + ((e >> 3) << 4) + lh * 8;
                ah[e] = (_Float16)__expf(s_alpha[bA][k] - mb);
            }
        }
        __syncthreads();
        // -- phase 3: D = A x exp(T), f32 accumulate on the WMMA pipe --
        v8f acc = {};
        #pragma unroll
        for (int kc = 0; kc < 8; ++kc) {
            v16h a = s_A[kc][lane];
            acc = __builtin_amdgcn_wmma_f32_16x16x32_f16(
                      false, a, false, Bt[kc], (short)0, acc, false, false);
        }
        // -- phase 4: alpha' = logit + M + log(acc), masked --
        {
            const int n = wave * 16 + nIdx;
            #pragma unroll
            for (int r = 0; r < 8; ++r) {
                const int m  = r + lh * 8;         // chain row in C/D layout
                const int gb = bb + m;
                const float lg = logits[(size_t)gb * SEQ * NUM_TAGS
                                        + (size_t)t * NUM_TAGS + n];
                const float nv = lg + s_M[m] + __logf(acc[r]);
                const int   mk = mask[gb * SEQ + t];
                s_alpha[m][n] = mk ? nv : s_alpha[m][n];
            }
            if (t + 1 < SEQ)
                __builtin_prefetch(&logits[(size_t)(bb + lh * 8) * SEQ * NUM_TAGS
                                           + (size_t)(t + 1) * NUM_TAGS + n], 0, 1);
        }
        __syncthreads();
    }

    // ---- final exact f32 logsumexp over tags per chain ----
    if (tid < 256) {
        const int b = tid >> 4, j0 = (tid & 15) << 4;
        float m = s_alpha[b][j0];
        #pragma unroll
        for (int j = 1; j < 16; ++j) m = fmaxf(m, s_alpha[b][j0 + j]);
        s_scr[b][tid & 15] = m;
    }
    __syncthreads();
    if (tid < BPB) {
        float m = s_scr[tid][0];
        #pragma unroll
        for (int j = 1; j < 16; ++j) m = fmaxf(m, s_scr[tid][j]);
        s_M[tid] = m;
    }
    __syncthreads();
    if (tid < 256) {
        const int b = tid >> 4, j0 = (tid & 15) << 4;
        float s = 0.f;
        #pragma unroll
        for (int j = 0; j < 16; ++j) s += __expf(s_alpha[b][j0 + j] - s_M[b]);
        s_scr[b][tid & 15] = s;
    }
    __syncthreads();
    if (tid < BPB) {
        float s = 0.f;
        #pragma unroll
        for (int j = 0; j < 16; ++j) s += s_scr[tid][j];
        log_den[bb + tid] = s_M[tid] + __logf(s);
    }
}

// ---------------------------------------------------------------------------
// Joint (numerator) score: one block per batch chain, full mask semantics.
// ---------------------------------------------------------------------------
__global__ __launch_bounds__(256)
void crf_joint_kernel(const float* __restrict__ logits,
                      const int*   __restrict__ tags,
                      const int*   __restrict__ mask,
                      const float* __restrict__ transitions,
                      float*       __restrict__ log_num)
{
    __shared__ float s_sum[256];
    __shared__ int   s_msk[256];
    const int b = blockIdx.x, tid = threadIdx.x;
    float acc = 0.f; int msum = 0;
    for (int t = tid; t < SEQ; t += 256) {
        const int mt = mask[b * SEQ + t];
        msum += mt;
        if (t < SEQ - 1) {
            const int tg  = tags[b * SEQ + t];
            const int tg1 = tags[b * SEQ + t + 1];
            acc += transitions[tg * NUM_TAGS + tg1] * (float)mask[b * SEQ + t + 1];
            acc += logits[(size_t)b * SEQ * NUM_TAGS + (size_t)t * NUM_TAGS + tg] * (float)mt;
        }
    }
    s_sum[tid] = acc; s_msk[tid] = msum;
    __syncthreads();
    for (int s = 128; s > 0; s >>= 1) {
        if (tid < s) { s_sum[tid] += s_sum[tid + s]; s_msk[tid] += s_msk[tid + s]; }
        __syncthreads();
    }
    if (tid == 0) {
        const int last_idx = s_msk[0] - 1;
        const int last_tag = tags[b * SEQ + last_idx];
        log_num[b] = s_sum[0]
            + logits[(size_t)b * SEQ * NUM_TAGS + (size_t)(SEQ - 1) * NUM_TAGS + last_tag]
              * (float)mask[b * SEQ + SEQ - 1];
    }
}

// ---------------------------------------------------------------------------
// Final scalar: sum_b (log_num[b] - log_den[b])
// ---------------------------------------------------------------------------
__global__ __launch_bounds__(64)
void crf_final_kernel(const float* __restrict__ num,
                      const float* __restrict__ den,
                      float* __restrict__ out)
{
    __shared__ float s[64];
    const int tid = threadIdx.x;
    s[tid] = num[tid] - den[tid];
    __syncthreads();
    for (int k = 32; k > 0; k >>= 1) {
        if (tid < k) s[tid] += s[tid + k];
        __syncthreads();
    }
    if (tid == 0) out[0] = s[0];
}

extern "C" void kernel_launch(void* const* d_in, const int* in_sizes, int n_in,
                              void* d_out, int out_size, void* d_ws, size_t ws_size,
                              hipStream_t stream) {
    const float* logits      = (const float*)d_in[0];
    const int*   tags        = (const int*)  d_in[1];
    const int*   mask        = (const int*)  d_in[2];
    const float* transitions = (const float*)d_in[3];
    float* out = (float*)d_out;
    float* num = (float*)d_ws;          // 64 floats
    float* den = num + BATCH;           // 64 floats

    crf_joint_kernel  <<<BATCH,       256,        0, stream>>>(logits, tags, mask, transitions, num);
    crf_forward_kernel<<<BATCH / BPB, FWD_THREADS, 0, stream>>>(logits, mask, transitions, den);
    crf_final_kernel  <<<1,           64,         0, stream>>>(num, den, out);
}
